// LocalAttention_14431090115032
// MI455X (gfx1250) — compile-verified, matrix operand order checked by reading, not run on hardware
//
#include <hip/hip_runtime.h>

// ---------------------------------------------------------------------------
// Local (banded) attention for gfx1250 (wave32, WMMA bf16 / f32 accumulate).
//   P0: pack x -> bf16; pack+transpose Wqkv, Wproj -> bf16 (N-major)
//   K1: qkv = x @ Wqkv        -> q/k/v scratch, [s][b][h][n][d] bf16
//   K2: fused banded softmax-attention -> attn scratch [b][n][c] bf16
//   K3: out = attn @ Wproj + b -> d_out f32
// GEMM waves compute 16x64 tiles: one A fragment feeds 4 WMMAs (2.5 b128
// loads per WMMA). All fragments are contiguous 16-byte loads; weights are
// pre-transposed so B fragments follow the same per-lane pattern as A.
// ---------------------------------------------------------------------------

typedef __attribute__((ext_vector_type(16))) __bf16          v16bf;
typedef __attribute__((ext_vector_type(8)))  __bf16          v8bf;
typedef __attribute__((ext_vector_type(8)))  float           v8f;
typedef __attribute__((ext_vector_type(4)))  unsigned short  v4us;

#define BATCH 4
#define NTOK  2048
#define DIM   512
#define HEADS 8
#define HD    64
#define WIN   8            // half-window: |i-j| <= 8

// float -> bf16 bits, round-to-nearest-even
__device__ __forceinline__ unsigned short f2bf_bits(float f) {
  union { float f; unsigned u; } c; c.f = f;
  unsigned r = (c.u + 0x7FFFu + ((c.u >> 16) & 1u)) >> 16;
  return (unsigned short)r;
}
__device__ __forceinline__ __bf16 bits2bf(unsigned short s) {
  union { unsigned short s; __bf16 b; } o; o.s = s; return o.b;
}

// Per-lane A/B fragment K offsets (ISA 7.12.2, 16-bit 16x32 / 32x16):
// element i<8 -> K = off+i; i>=8 -> K = off+16+(i-8)  (off = hi*8).
// Two contiguous 16B runs -> two b128 loads + register concat.
__device__ __forceinline__ v16bf frag_ld(const unsigned short* p) {
  const v8bf a0 = *(const v8bf*)(p);
  const v8bf a1 = *(const v8bf*)(p + 16);
  return __builtin_shufflevector(a0, a1, 0, 1, 2, 3, 4, 5, 6, 7,
                                 8, 9, 10, 11, 12, 13, 14, 15);
}

// ---------------------------------------------------------------------------
// P0a: elementwise f32 -> bf16 (vectorized x4)
// ---------------------------------------------------------------------------
__global__ void convert_bf16(const float* __restrict__ in,
                             unsigned short* __restrict__ out, int n4) {
  const int i = blockIdx.x * blockDim.x + threadIdx.x;
  if (i >= n4) return;
  const float4 f = ((const float4*)in)[i];
  v4us o;
  o.x = f2bf_bits(f.x); o.y = f2bf_bits(f.y);
  o.z = f2bf_bits(f.z); o.w = f2bf_bits(f.w);
  ((v4us*)out)[i] = o;
}

// ---------------------------------------------------------------------------
// P0b: f32 [R x C] row-major -> bf16 [C x R] (transpose + convert)
// ---------------------------------------------------------------------------
__global__ void convert_transpose(const float* __restrict__ W,
                                  unsigned short* __restrict__ WT,
                                  int R, int C) {
  const int idx = blockIdx.x * blockDim.x + threadIdx.x;
  if (idx >= R * C) return;
  const int r = idx / C, c = idx % C;
  WT[(size_t)c * R + r] = f2bf_bits(W[idx]);
}

// ---------------------------------------------------------------------------
// K1: qkv = xb[8192x512] @ WqkvT[1536x512]^T.  16x64 tile per wave.
// Scatter to [s][b][h][n][d] bf16.
// ---------------------------------------------------------------------------
__global__ void qkv_gemm(const unsigned short* __restrict__ xb,
                         const unsigned short* __restrict__ WqkvT,
                         unsigned short* __restrict__ qkv_ws) {
  const int m0   = blockIdx.x * 16;
  const int n0   = blockIdx.y * 64;
  const int lane = threadIdx.x & 31;
  const int lo   = lane & 15;
  const int hi   = lane >> 4;

  const unsigned short* arow = xb + (size_t)(m0 + lo) * DIM + hi * 8;
  const unsigned short* brow[4];
#pragma unroll
  for (int c = 0; c < 4; ++c)
    brow[c] = WqkvT + (size_t)(n0 + c * 16 + lo) * DIM + hi * 8;

  v8f acc[4] = {{}, {}, {}, {}};
  for (int k0 = 0; k0 < DIM; k0 += 32) {
    __builtin_prefetch(arow + k0 + 128, 0, 1);       // global_prefetch_b8
    const v16bf a = frag_ld(arow + k0);
#pragma unroll
    for (int c = 0; c < 4; ++c) {
      const v16bf b = frag_ld(brow[c] + k0);
      acc[c] = __builtin_amdgcn_wmma_f32_16x16x32_bf16(
          false, a, false, b, (short)0, acc[c], false, false);
    }
  }

  // scatter: column -> (s, h, d); row -> (b, n)
  const int bb = m0 / NTOK;              // 16-row tile never crosses batch
  const int nt0 = m0 % NTOK;
#pragma unroll
  for (int c = 0; c < 4; ++c) {
    const int col = n0 + c * 16 + lo;
    const int s   = col / DIM;
    const int h   = (col % DIM) / HD;
    const int d   = col % HD;
    const size_t base =
        (((size_t)s * BATCH + bb) * HEADS + h) * NTOK * HD + d;
#pragma unroll
    for (int r = 0; r < 8; ++r)
      qkv_ws[base + (size_t)(nt0 + r + hi * 8) * HD] = f2bf_bits(acc[c][r]);
  }
}

// ---------------------------------------------------------------------------
// K2: fused banded attention. One wave per (b, h, 16-query tile).
// Key range for a 16-row query tile is exactly 32 wide: [q0-8, q0+23].
// ---------------------------------------------------------------------------
__global__ void local_attn(const unsigned short* __restrict__ qkv_ws,
                           unsigned short* __restrict__ attn_ws) {
  const int tile = blockIdx.x;
  const int qt   = tile % (NTOK / 16);
  const int bh   = tile / (NTOK / 16);       // b*HEADS + h
  const int h    = bh % HEADS;
  const int b    = bh / HEADS;
  const int lane = threadIdx.x & 31;
  const int lo   = lane & 15;
  const int hi   = lane >> 4;
  const int q0   = qt * 16;
  const int jlo  = q0 - WIN;

  const unsigned short* qp = qkv_ws + ((size_t)(0 * BATCH * HEADS) + bh) * NTOK * HD;
  const unsigned short* kp = qkv_ws + ((size_t)(1 * BATCH * HEADS) + bh) * NTOK * HD;
  const unsigned short* vp = qkv_ws + ((size_t)(2 * BATCH * HEADS) + bh) * NTOK * HD;

  __shared__ unsigned short Vlds[32 * HD];   // V tile [key][d]
  __shared__ unsigned short Plds[16 * 32];   // P tile [row][key]

  // ---- stage V tile (32 keys x 64 dims) into LDS, wide & coalesced ----
  {
    const int jg = jlo + lane;
    const int jc = jg < 0 ? 0 : (jg >= NTOK ? NTOK - 1 : jg);
    const unsigned short* src = vp + (size_t)jc * HD;
#pragma unroll
    for (int d0 = 0; d0 < HD; d0 += 8)
      *(v8bf*)&Vlds[lane * HD + d0] = *(const v8bf*)(src + d0);
  }

  // ---- S = Q @ K^T : two 16x16 key tiles, K-dim 64 = 2 x wmma(K=32) ----
  const unsigned short* qrow = qp + (size_t)(q0 + lo) * HD + hi * 8;
  v16bf qa[2];
#pragma unroll
  for (int d0 = 0; d0 < HD; d0 += 32) qa[d0 / 32] = frag_ld(qrow + d0);

  v8f s_acc[2] = {{}, {}};
#pragma unroll
  for (int t = 0; t < 2; ++t) {
    const int jg = jlo + t * 16 + lo;
    const int jc = jg < 0 ? 0 : (jg >= NTOK ? NTOK - 1 : jg);
    const unsigned short* krow = kp + (size_t)jc * HD + hi * 8;
#pragma unroll
    for (int d0 = 0; d0 < HD; d0 += 32) {
      const v16bf kb = frag_ld(krow + d0);
      s_acc[t] = __builtin_amdgcn_wmma_f32_16x16x32_bf16(
          false, qa[d0 / 32], false, kb, (short)0, s_acc[t], false, false);
    }
  }

  // ---- band mask + softmax (rows live in 16-lane half-wave groups) ----
  const float scale = 0.125f;                // hd^-0.5
#pragma unroll
  for (int r = 0; r < 8; ++r) {
    const int qi = q0 + r + hi * 8;
    const int j0 = jlo + lo;
    const int j1 = jlo + 16 + lo;
    const int d0 = qi - j0, d1 = qi - j1;
    const bool ok0 = (j0 >= 0) && (j0 < NTOK) && (d0 <= WIN) && (d0 >= -WIN);
    const bool ok1 = (j1 >= 0) && (j1 < NTOK) && (d1 <= WIN) && (d1 >= -WIN);
    float s0 = ok0 ? s_acc[0][r] * scale : -3.0e38f;
    float s1 = ok1 ? s_acc[1][r] * scale : -3.0e38f;
    float rmax = fmaxf(s0, s1);
#pragma unroll
    for (int off = 1; off < 16; off <<= 1)
      rmax = fmaxf(rmax, __shfl_xor(rmax, off, 16));
    float e0 = ok0 ? __expf(s0 - rmax) : 0.f;
    float e1 = ok1 ? __expf(s1 - rmax) : 0.f;
    float rsum = e0 + e1;
#pragma unroll
    for (int off = 1; off < 16; off <<= 1)
      rsum += __shfl_xor(rsum, off, 16);
    const float inv = 1.f / rsum;            // diagonal always in band
    const int m = r + hi * 8;
    Plds[m * 32 + lo]      = f2bf_bits(e0 * inv);
    Plds[m * 32 + 16 + lo] = f2bf_bits(e1 * inv);
  }
  __syncthreads();

  // ---- P fragment: A-layout runs are contiguous in the LDS row ----
  const v16bf pa = frag_ld(&Plds[lo * 32 + hi * 8]);   // 2 x ds_load_b128

  // ---- O = P[16x32] @ V[32x64] : one wmma(K=32) per 16-col chunk ----
  const size_t obase = ((size_t)b * NTOK + q0) * DIM + h * HD;
#pragma unroll
  for (int c = 0; c < 4; ++c) {
    v16bf vb;
#pragma unroll
    for (int i = 0; i < 16; ++i) {
      const int k = (i < 8 ? i : i + 8) + hi * 8;      // key row 0..31
      vb[i] = bits2bf(Vlds[k * HD + c * 16 + lo]);
    }
    v8f o = {};
    o = __builtin_amdgcn_wmma_f32_16x16x32_bf16(false, pa, false, vb,
                                                (short)0, o, false, false);
#pragma unroll
    for (int r = 0; r < 8; ++r)
      attn_ws[obase + (size_t)(r + hi * 8) * DIM + c * 16 + lo] =
          f2bf_bits(o[r]);
  }
}

// ---------------------------------------------------------------------------
// K3: out = attn[8192x512](bf16) @ WprojT[512x512]^T + bproj  (f32 out).
// 16x64 tile per wave, A-fragment reuse across 4 WMMAs.
// ---------------------------------------------------------------------------
__global__ void proj_gemm(const unsigned short* __restrict__ attn_ws,
                          const unsigned short* __restrict__ WprojT,
                          const float* __restrict__ bproj,
                          float* __restrict__ out) {
  const int m0   = blockIdx.x * 16;
  const int n0   = blockIdx.y * 64;
  const int lane = threadIdx.x & 31;
  const int lo   = lane & 15;
  const int hi   = lane >> 4;

  const unsigned short* arow = attn_ws + (size_t)(m0 + lo) * DIM + hi * 8;
  const unsigned short* brow[4];
#pragma unroll
  for (int c = 0; c < 4; ++c)
    brow[c] = WprojT + (size_t)(n0 + c * 16 + lo) * DIM + hi * 8;

  v8f acc[4] = {{}, {}, {}, {}};
  for (int k0 = 0; k0 < DIM; k0 += 32) {
    __builtin_prefetch(arow + k0 + 128, 0, 1);
    const v16bf a = frag_ld(arow + k0);
#pragma unroll
    for (int c = 0; c < 4; ++c) {
      const v16bf b = frag_ld(brow[c] + k0);
      acc[c] = __builtin_amdgcn_wmma_f32_16x16x32_bf16(
          false, a, false, b, (short)0, acc[c], false, false);
    }
  }
#pragma unroll
  for (int c = 0; c < 4; ++c) {
    const float bias = bproj[n0 + c * 16 + lo];
#pragma unroll
    for (int r = 0; r < 8; ++r)
      out[(size_t)(m0 + r + hi * 8) * DIM + n0 + c * 16 + lo] =
          acc[c][r] + bias;
  }
}

// ---------------------------------------------------------------------------
extern "C" void kernel_launch(void* const* d_in, const int* in_sizes, int n_in,
                              void* d_out, int out_size, void* d_ws,
                              size_t ws_size, hipStream_t stream) {
  const float* x     = (const float*)d_in[0];  // [4,2048,512]
  const float* Wqkv  = (const float*)d_in[1];  // [512,1536]
  const float* Wproj = (const float*)d_in[2];  // [512,512]
  const float* bproj = (const float*)d_in[3];  // [512]
  float*       out   = (float*)d_out;          // [4,2048,512]

  const int BN = BATCH * NTOK;                 // 8192

  // workspace layout (ushort units)
  unsigned short* ws      = (unsigned short*)d_ws;
  unsigned short* qkv_ws  = ws;                                        // 12.58M
  unsigned short* attn_ws = qkv_ws  + (size_t)3 * BATCH * HEADS * NTOK * HD;
  unsigned short* xb      = attn_ws + (size_t)BN * DIM;                //  4.19M
  unsigned short* WqkvT   = xb      + (size_t)BN * DIM;                //  0.79M
  unsigned short* WprojT  = WqkvT   + (size_t)3 * DIM * DIM;           //  0.26M

  // P0: conversions / transposes (bandwidth-trivial)
  {
    const int n4 = BN * DIM / 4;
    convert_bf16<<<dim3((n4 + 255) / 256), 256, 0, stream>>>(x, xb, n4);
    const int nq = DIM * 3 * DIM;
    convert_transpose<<<dim3((nq + 255) / 256), 256, 0, stream>>>(
        Wqkv, WqkvT, DIM, 3 * DIM);
    const int np = DIM * DIM;
    convert_transpose<<<dim3((np + 255) / 256), 256, 0, stream>>>(
        Wproj, WprojT, DIM, DIM);
  }

  qkv_gemm<<<dim3(BN / 16, (3 * DIM) / 64), 32, 0, stream>>>(xb, WqkvT,
                                                             qkv_ws);
  local_attn<<<dim3(BATCH * HEADS * (NTOK / 16)), 32, 0, stream>>>(qkv_ws,
                                                                   attn_ws);
  proj_gemm<<<dim3(BN / 16, DIM / 64), 32, 0, stream>>>(attn_ws, WprojT,
                                                        bproj, out);
}
